// SemanticModule_29858612642627
// MI455X (gfx1250) — compile-verified, matrix operand order checked by reading, not run on hardware
//
#include <hip/hip_runtime.h>
#include <stdint.h>

// Problem constants (fixed by the reference's setup_inputs()).
#define NN       100000
#define E_TEMP   100000
#define E_INTER  3200000

typedef __attribute__((ext_vector_type(16))) __bf16 v16bf;
typedef __attribute__((ext_vector_type(8)))  float  v8f;

// fp32 -> bf16 with round-to-nearest-even, packed pair into one dword.
__device__ __forceinline__ uint32_t pack_bf16x2(float a, float b) {
  uint32_t ua = __float_as_uint(a);
  uint32_t ub = __float_as_uint(b);
  ua = (ua + 0x7FFFu + ((ua >> 16) & 1u)) >> 16;
  ub = (ub + 0x7FFFu + ((ub >> 16) & 1u)) >> 16;
  return (ua & 0xFFFFu) | (ub << 16);
}

__global__ void k_zero(float* __restrict__ p, int n) {
  int i = blockIdx.x * blockDim.x + threadIdx.x;
  if (i < n) p[i] = 0.f;
}

__global__ void k_count(const int* __restrict__ dst, int e, float* __restrict__ cnt) {
  int i = blockIdx.x * blockDim.x + threadIdx.x;
  if (i < e) atomicAdd(cnt + dst[i], 1.f);
}

__global__ void k_invert(float* __restrict__ cnt, int n) {
  int i = blockIdx.x * blockDim.x + threadIdx.x;
  if (i < n) cnt[i] = 1.f / fmaxf(cnt[i], 1.f);
}

// Pack one weight matrix W[K x M] (fp32, K zero-padded to 32) into the wave32
// bf16 B-operand register layout for v_wmma_f32_16x16x32_bf16.
// Lane-major output layout: pw[tile*256 + lane*8 + v], so the GEMM kernel
// fetches each lane's 8 dwords with two global_load_b128.
// K-index for (v, half): v<4 -> 8*half + 2v ; v>=4 -> 16 + 8*half + 2(v-4).
__global__ void k_pack_w(const float* __restrict__ W, int Kact, int M,
                         uint32_t* __restrict__ out) {
  int tid = blockIdx.x * blockDim.x + threadIdx.x;
  int total = (M >> 4) * 256;
  if (tid >= total) return;
  int tile = tid >> 8;
  int lane = (tid >> 3) & 31;
  int v    = tid & 7;
  int half = lane >> 4;
  int col  = (tile << 4) + (lane & 15);
  int k0 = ((v < 4) ? (2 * v) : (16 + 2 * (v - 4))) + half * 8;
  float f0 = (k0     < Kact) ? W[k0 * M + col]       : 0.f;
  float f1 = (k0 + 1 < Kact) ? W[(k0 + 1) * M + col] : 0.f;
  out[tid] = pack_bf16x2(f0, f1);
}

// One wave computes one 16-row tile of X[N x KACT] against NW pre-packed
// weight matrices (each K zero-padded to 32, M = NCT*16 cols), fully unrolled:
// NW*NCT back-to-back v_wmma_f32_16x16x32_bf16 per wave.
// Weight index w==2 (the root/self transform) additionally gets +bias.
template <int NW, int NCT, int KACT>
__global__ __launch_bounds__(256) void k_gemm_wmma(
    const float* __restrict__ X, const uint32_t* __restrict__ pw,
    float* o0, float* o1, float* o2, float* o3,
    const float* __restrict__ bias, int nRowTiles) {
  int wave = (int)((blockIdx.x * blockDim.x + threadIdx.x) >> 5);
  if (wave >= nRowTiles) return;          // wave-uniform: EXEC stays all-ones
  int lane = threadIdx.x & 31;
  int half = lane >> 4;
  int row  = (wave << 4) + (lane & 15);
  constexpr int M = NCT * 16;

  // A operand: row `row` of X, bf16-converted, 16-bit 16x32 A layout.
  union { uint32_t u[8]; v16bf vec; } a;
  if constexpr (KACT == 32) {
    // Lane's data = two contiguous 32B chunks: K in [8h, 8h+8) and [16+8h, 16+8h+8).
    const float4* x0 = (const float4*)(X + (size_t)row * 32 + half * 8);
    const float4* x1 = (const float4*)(X + (size_t)row * 32 + 16 + half * 8);
    float4 f0 = x0[0], f1 = x0[1], g0 = x1[0], g1 = x1[1];
    a.u[0] = pack_bf16x2(f0.x, f0.y); a.u[1] = pack_bf16x2(f0.z, f0.w);
    a.u[2] = pack_bf16x2(f1.x, f1.y); a.u[3] = pack_bf16x2(f1.z, f1.w);
    a.u[4] = pack_bf16x2(g0.x, g0.y); a.u[5] = pack_bf16x2(g0.z, g0.w);
    a.u[6] = pack_bf16x2(g1.x, g1.y); a.u[7] = pack_bf16x2(g1.z, g1.w);
  } else {
#pragma unroll
    for (int v = 0; v < 8; ++v) {
      int k0 = ((v < 4) ? (2 * v) : (16 + 2 * (v - 4))) + half * 8;
      float f0 = (k0     < KACT) ? X[row * KACT + k0]     : 0.f;
      float f1 = (k0 + 1 < KACT) ? X[row * KACT + k0 + 1] : 0.f;
      a.u[v] = pack_bf16x2(f0, f1);
    }
  }

  float* outs[4] = {o0, o1, o2, o3};
  const int col_lo = lane & 15;
#pragma unroll
  for (int w = 0; w < NW; ++w) {
    float* out = outs[w];
#pragma unroll
    for (int t = 0; t < NCT; ++t) {
      union { uint32_t u[8]; v16bf vec; } bm;
      const uint4* bp = (const uint4*)(pw + ((w * NCT + t) << 8) + (lane << 3));
      uint4 b0 = bp[0], b1 = bp[1];
      bm.u[0] = b0.x; bm.u[1] = b0.y; bm.u[2] = b0.z; bm.u[3] = b0.w;
      bm.u[4] = b1.x; bm.u[5] = b1.y; bm.u[6] = b1.z; bm.u[7] = b1.w;

      v8f c = {};
      c = __builtin_amdgcn_wmma_f32_16x16x32_bf16(
              /*neg_a=*/false, a.vec, /*neg_b=*/false, bm.vec,
              /*c_mod=*/(short)0, c, /*reuse_a=*/false, /*reuse_b=*/false);

      int col = (t << 4) + col_lo;
      float badd = (w == 2) ? bias[col] : 0.f;
#pragma unroll
      for (int r = 0; r < 8; ++r) {
        int rr = (wave << 4) + (half << 3) + r;   // C/D layout: M = r + 8*half
        out[(size_t)rr * M + col] = c[r] + badd;
      }
    }
  }
}

// Edge scatter: acc[dst] += xs[src] * (inv ? inv[dst] : 1).
// One thread moves 4 consecutive channels (float4 load, 4 f32 atomics to L2).
__global__ void k_scatter(const int* __restrict__ src, const int* __restrict__ dst,
                          int E, int m4shift, const float* __restrict__ xs,
                          float* __restrict__ acc, const float* __restrict__ inv,
                          int M) {
  int tid = blockIdx.x * blockDim.x + threadIdx.x;
  int total = E << m4shift;
  if (tid >= total) return;
  int e  = tid >> m4shift;
  int c4 = (tid & ((1 << m4shift) - 1)) << 2;
  int s = src[e], d = dst[e];
  float4 val = *(const float4*)(xs + s * M + c4);
  float scale = inv ? inv[d] : 1.f;
  float* p = acc + d * M + c4;
  atomicAdd(p + 0, val.x * scale);
  atomicAdd(p + 1, val.y * scale);
  atomicAdd(p + 2, val.z * scale);
  atomicAdd(p + 3, val.w * scale);
}

// out = relu(acc) + add   (add may be null; in-place out==add is safe).
__global__ void k_finalize(const float* __restrict__ acc, const float* __restrict__ add,
                           float* __restrict__ out, int n4) {
  int i = blockIdx.x * blockDim.x + threadIdx.x;
  if (i >= n4) return;
  float4 a = ((const float4*)acc)[i];
  float4 r;
  r.x = fmaxf(a.x, 0.f); r.y = fmaxf(a.y, 0.f);
  r.z = fmaxf(a.z, 0.f); r.w = fmaxf(a.w, 0.f);
  if (add) {
    float4 h = ((const float4*)add)[i];
    r.x += h.x; r.y += h.y; r.z += h.z; r.w += h.w;
  }
  ((float4*)out)[i] = r;
}

extern "C" void kernel_launch(void* const* d_in, const int* in_sizes, int n_in,
                              void* d_out, int out_size, void* d_ws, size_t ws_size,
                              hipStream_t stream) {
  const float* x        = (const float*)d_in[0];
  const int*   ei_temp  = (const int*)d_in[1];
  const int*   ei_inter = (const int*)d_in[2];
  const float* head_Wt  = (const float*)d_in[3];
  const float* head_Wi  = (const float*)d_in[4];
  const float* head_Wr  = (const float*)d_in[5];
  const float* head_b   = (const float*)d_in[6];
  const float* blk_Wt   = (const float*)d_in[7];
  const float* blk_Wi   = (const float*)d_in[8];
  const float* blk_Wr   = (const float*)d_in[9];
  const float* blk_b    = (const float*)d_in[10];
  const float* last_Wt  = (const float*)d_in[11];
  const float* last_Wi  = (const float*)d_in[12];
  const float* last_Wr  = (const float*)d_in[13];
  const float* last_b   = (const float*)d_in[14];
  const float* last_proj= (const float*)d_in[15];
  float* out = (float*)d_out;

  // Workspace carve-up (~90 MB).
  float* ws  = (float*)d_ws;
  float* inv = ws;                          // NN
  float* h   = inv + NN;                    // NN*32 hidden features
  float* xt  = h  + (size_t)NN * 32;        // NN*64 (max M)
  float* xi  = xt + (size_t)NN * 64;        // NN*64
  float* acc = xi + (size_t)NN * 64;        // NN*64
  uint32_t* pw      = (uint32_t*)(acc + (size_t)NN * 64);
  uint32_t* pw_head = pw;                   // 3 mats * 2 tiles * 256
  uint32_t* pw_blk  = pw_head + 3 * 512;    // 3 layers * 3 mats * 2 tiles * 256
  uint32_t* pw_last = pw_blk  + 3 * 3 * 512;// 4 mats * 4 tiles * 256

  const int* t_src = ei_temp;  const int* t_dst = ei_temp  + E_TEMP;
  const int* i_src = ei_inter; const int* i_dst = ei_inter + E_INTER;

  const int B = 256;
  auto g = [](int n, int b) { return (n + b - 1) / b; };

  // 1/max(deg,1) for the 'intersects' mean aggregation (edge-level scaling).
  k_zero  <<<g(NN, B),      B, 0, stream>>>(inv, NN);
  k_count <<<g(E_INTER, B), B, 0, stream>>>(i_dst, E_INTER, inv);
  k_invert<<<g(NN, B),      B, 0, stream>>>(inv, NN);

  // Pre-pack all weights into WMMA B-operand bf16 layout (K zero-padded to 32).
  k_pack_w<<<2, B, 0, stream>>>(head_Wt, 6, 32, pw_head + 0 * 512);
  k_pack_w<<<2, B, 0, stream>>>(head_Wi, 6, 32, pw_head + 1 * 512);
  k_pack_w<<<2, B, 0, stream>>>(head_Wr, 6, 32, pw_head + 2 * 512);
  for (int i = 0; i < 3; ++i) {
    k_pack_w<<<2, B, 0, stream>>>(blk_Wt + i * 1024, 32, 32, pw_blk + i * 1536 + 0 * 512);
    k_pack_w<<<2, B, 0, stream>>>(blk_Wi + i * 1024, 32, 32, pw_blk + i * 1536 + 1 * 512);
    k_pack_w<<<2, B, 0, stream>>>(blk_Wr + i * 1024, 32, 32, pw_blk + i * 1536 + 2 * 512);
  }
  k_pack_w<<<4, B, 0, stream>>>(last_Wt,   32, 64, pw_last + 0 * 1024);
  k_pack_w<<<4, B, 0, stream>>>(last_Wi,   32, 64, pw_last + 1 * 1024);
  k_pack_w<<<4, B, 0, stream>>>(last_Wr,   32, 64, pw_last + 2 * 1024);
  k_pack_w<<<4, B, 0, stream>>>(last_proj, 32, 64, pw_last + 3 * 1024);

  const int nRowTiles  = NN / 16;                 // 6250 (exact)
  const int gemmBlocks = g(nRowTiles * 32, B);

  // ---- head: 6 -> 32 ----
  k_gemm_wmma<3, 2, 6><<<gemmBlocks, B, 0, stream>>>(
      x, pw_head, xt, xi, acc, nullptr, head_b, nRowTiles);
  k_scatter<<<g(E_TEMP  * 8, B), B, 0, stream>>>(t_src, t_dst, E_TEMP,  3, xt, acc, nullptr, 32);
  k_scatter<<<g(E_INTER * 8, B), B, 0, stream>>>(i_src, i_dst, E_INTER, 3, xi, acc, inv,     32);
  k_finalize<<<g(NN * 8, B), B, 0, stream>>>(acc, nullptr, h, NN * 8);

  // ---- 3 residual blocks: 32 -> 32, h = relu(conv(h)) + h ----
  for (int i = 0; i < 3; ++i) {
    k_gemm_wmma<3, 2, 32><<<gemmBlocks, B, 0, stream>>>(
        h, pw_blk + i * 1536, xt, xi, acc, nullptr, blk_b + i * 32, nRowTiles);
    k_scatter<<<g(E_TEMP  * 8, B), B, 0, stream>>>(t_src, t_dst, E_TEMP,  3, xt, acc, nullptr, 32);
    k_scatter<<<g(E_INTER * 8, B), B, 0, stream>>>(i_src, i_dst, E_INTER, 3, xi, acc, inv,     32);
    k_finalize<<<g(NN * 8, B), B, 0, stream>>>(acc, h, h, NN * 8);
  }

  // ---- last: 32 -> 64, residual via h @ last_proj (4th GEMM into d_out) ----
  k_gemm_wmma<4, 4, 32><<<gemmBlocks, B, 0, stream>>>(
      h, pw_last, xt, xi, acc, out, last_b, nRowTiles);
  k_scatter<<<g(E_TEMP  * 16, B), B, 0, stream>>>(t_src, t_dst, E_TEMP,  4, xt, acc, nullptr, 64);
  k_scatter<<<g(E_INTER * 16, B), B, 0, stream>>>(i_src, i_dst, E_INTER, 4, xi, acc, inv,     64);
  k_finalize<<<g(NN * 16, B), B, 0, stream>>>(acc, out, out, NN * 16);
}
